// NexusNet_4853313045170
// MI455X (gfx1250) — compile-verified, matrix operand order checked by reading, not run on hardware
//
#include <hip/hip_runtime.h>
#include <math.h>

#define N_PLANE 200000
#define N_NEXUS 50000
#define N_EDGE  400000

typedef __attribute__((ext_vector_type(16))) _Float16 v16h;
typedef __attribute__((ext_vector_type(8)))  _Float16 v8h;
typedef __attribute__((ext_vector_type(8)))  float    v8f;

// Load a 16x32-f16 WMMA A/B fragment for this lane from a row-major [rows][ldk]
// f16 tile. p points at row_base + k0 + 8*h. Per ISA 7.12.2:
//   packed elem p<8  -> K = k0 + 8h + p
//   packed elem p>=8 -> K = k0 + 16 + 8h + (p-8)
// => two contiguous 16-byte loads.
__device__ __forceinline__ v16h load_frag(const _Float16* p) {
  v8h lo = *(const v8h*)(p);
  v8h hi = *(const v8h*)(p + 16);
  v16h f;
#pragma unroll
  for (int i = 0; i < 8; ++i) { f[i] = lo[i]; f[i + 8] = hi[i]; }
  return f;
}

__device__ __forceinline__ v8f wmma16(v16h a, v16h b, v8f c) {
  return __builtin_amdgcn_wmma_f32_16x16x32_f16(false, a, false, b, (short)0, c,
                                                false, false);
}

// W [K][N] f32 (row-major, as in the reference) -> WT [N][K] f16
__global__ void wt_f16_kernel(const float* __restrict__ W,
                              _Float16* __restrict__ WT, int K, int N) {
  int tid = blockIdx.x * blockDim.x + threadIdx.x;
  if (tid >= K * N) return;
  int k = tid / N, n = tid - k * N;
  WT[n * K + k] = (_Float16)W[tid];
}

// nexus_up: n_up[e1, col0 + k] += x[e0, k]  (one thread per edge-feature)
__global__ void up_scatter_kernel(const float* __restrict__ x,
                                  const int* __restrict__ edge,
                                  float* __restrict__ n_up, int col0) {
  long tid = (long)blockIdx.x * blockDim.x + threadIdx.x;
  if (tid >= (long)N_EDGE * 64) return;
  int e = (int)(tid >> 6);
  int k = (int)(tid & 63);
  int src = edge[e];
  int dst = edge[N_EDGE + e];
  atomicAdd(&n_up[(size_t)dst * 192 + col0 + k], x[(size_t)src * 64 + k]);
}

__global__ void count_kernel(const int* __restrict__ edge,
                             float* __restrict__ cnt) {
  int e = blockIdx.x * blockDim.x + threadIdx.x;
  if (e < N_EDGE) atomicAdd(&cnt[edge[e]], 1.0f);
}

// n = tanh(tanh(n_up @ W1 + b1) @ W2 + b2), one wave per 16 nexus rows.
__global__ __launch_bounds__(128) void nexus_mlp_kernel(
    const float* __restrict__ n_up, const _Float16* __restrict__ W1T,
    const float* __restrict__ b1, const _Float16* __restrict__ W2T,
    const float* __restrict__ b2, float* __restrict__ n_out) {
  __shared__ __align__(16) _Float16 aT[4][16 * 192];
  __shared__ __align__(16) _Float16 hT[4][16 * 64];
  const int wave = threadIdx.x >> 5, lane = threadIdx.x & 31;
  const int r = lane & 15, h = lane >> 4;
  const int tile = blockIdx.x * 4 + wave;
  if (tile >= N_NEXUS / 16) return;  // 50000/16 = 3125 exact tiles
  const int row0 = tile * 16;
  _Float16* A = &aT[wave][0];
  _Float16* H = &hT[wave][0];

  // Stage 16x192 f32 -> f16 LDS tile (row-major, K contiguous)
  for (int m = 0; m < 16; ++m)
    for (int k = lane; k < 192; k += 32)
      A[m * 192 + k] = (_Float16)n_up[(size_t)(row0 + m) * 192 + k];

  // GEMM1: [16x192] @ [192x64], K-loop of 6, 4 n-tiles
  v8f c[4] = {};
#pragma unroll
  for (int k0 = 0; k0 < 192; k0 += 32) {
    v16h a = load_frag(&A[r * 192 + k0 + 8 * h]);
#pragma unroll
    for (int t = 0; t < 4; ++t) {
      v16h b = load_frag(&W1T[(size_t)(t * 16 + r) * 192 + k0 + 8 * h]);
      c[t] = wmma16(a, b, c[t]);
    }
  }
  // bias + tanh, write hidden back to LDS row-major for GEMM2 A-fragments
#pragma unroll
  for (int t = 0; t < 4; ++t) {
    float bias = b1[t * 16 + r];
#pragma unroll
    for (int i = 0; i < 8; ++i)
      H[(i + 8 * h) * 64 + t * 16 + r] = (_Float16)tanhf(c[t][i] + bias);
  }

  // GEMM2: [16x64] @ [64x64]
  v8f d[4] = {};
#pragma unroll
  for (int k0 = 0; k0 < 64; k0 += 32) {
    v16h a = load_frag(&H[r * 64 + k0 + 8 * h]);
#pragma unroll
    for (int t = 0; t < 4; ++t) {
      v16h b = load_frag(&W2T[(size_t)(t * 16 + r) * 64 + k0 + 8 * h]);
      d[t] = wmma16(a, b, d[t]);
    }
  }
#pragma unroll
  for (int t = 0; t < 4; ++t) {
    float bias = b2[t * 16 + r];
#pragma unroll
    for (int i = 0; i < 8; ++i)
      n_out[(size_t)(row0 + i + 8 * h) * 64 + t * 16 + r] =
          tanhf(d[t][i] + bias);
  }
}

// NexusDown: gate = sigmoid([xi||nj] @ Wd + bd); scatter-sum gate*nj over e0.
// One wave per 16 edges; A tile = 16x128 f16 in LDS, nj kept in f32 for the
// gated product; output accumulated with f32 atomics (mean finalized later).
__global__ __launch_bounds__(128) void nexus_down_kernel(
    const float* __restrict__ x, const float* __restrict__ nfeat,
    const int* __restrict__ edge, const _Float16* __restrict__ WdT,
    const float* __restrict__ bd, float* __restrict__ out) {
  __shared__ __align__(16) _Float16 aT[4][16 * 128];
  __shared__ __align__(16) float njT[4][16 * 64];
  __shared__ int e0T[4][16];
  const int wave = threadIdx.x >> 5, lane = threadIdx.x & 31;
  const int r = lane & 15, h = lane >> 4;
  const int tile = blockIdx.x * 4 + wave;  // 400000/16 = 25000 exact tiles
  const int base = tile * 16;
  _Float16* A = &aT[wave][0];
  float* NJ = &njT[wave][0];
  int* E0 = &e0T[wave][0];

  for (int m = 0; m < 16; ++m) {
    int e0 = edge[base + m];
    int e1 = edge[N_EDGE + base + m];
    if (lane == 0) E0[m] = e0;
    for (int k = lane; k < 64; k += 32) {
      A[m * 128 + k] = (_Float16)x[(size_t)e0 * 64 + k];
      float v = nfeat[(size_t)e1 * 64 + k];
      A[m * 128 + 64 + k] = (_Float16)v;
      NJ[m * 64 + k] = v;
    }
  }

  v8f c[4] = {};
#pragma unroll
  for (int k0 = 0; k0 < 128; k0 += 32) {
    v16h a = load_frag(&A[r * 128 + k0 + 8 * h]);
#pragma unroll
    for (int t = 0; t < 4; ++t) {
      v16h b = load_frag(&WdT[(size_t)(t * 16 + r) * 128 + k0 + 8 * h]);
      c[t] = wmma16(a, b, c[t]);
    }
  }

#pragma unroll
  for (int t = 0; t < 4; ++t) {
    const int n = t * 16 + r;
    float bias = bd[n];
#pragma unroll
    for (int i = 0; i < 8; ++i) {
      int m = i + 8 * h;  // D layout: VGPR i -> M = i + 8h, lane -> N = r
      float g = 1.0f / (1.0f + expf(-(c[t][i] + bias)));
      atomicAdd(&out[(size_t)E0[m] * 64 + n], g * NJ[m * 64 + n]);
    }
  }
}

// scatter_mean finalize: out /= max(count, 1)
__global__ void finalize_kernel(float* __restrict__ out,
                                const float* __restrict__ cnt) {
  long tid = (long)blockIdx.x * blockDim.x + threadIdx.x;
  if (tid >= 3L * N_PLANE * 64) return;
  long node = tid >> 6;  // plane-major [3*N_PLANE], matches out layout
  out[tid] = out[tid] / fmaxf(cnt[node], 1.0f);
}

extern "C" void kernel_launch(void* const* d_in, const int* in_sizes, int n_in,
                              void* d_out, int out_size, void* d_ws,
                              size_t ws_size, hipStream_t stream) {
  (void)in_sizes; (void)n_in; (void)out_size; (void)ws_size;
  const float* x_u = (const float*)d_in[0];
  const float* x_v = (const float*)d_in[1];
  const float* x_y = (const float*)d_in[2];
  // d_in[3] (nexus features) only contributes shape in the reference
  const int* edge_u = (const int*)d_in[4];
  const int* edge_v = (const int*)d_in[5];
  const int* edge_y = (const int*)d_in[6];
  const float* W1 = (const float*)d_in[7];
  const float* b1 = (const float*)d_in[8];
  const float* W2 = (const float*)d_in[9];
  const float* b2 = (const float*)d_in[10];
  const float* Wd[3] = {(const float*)d_in[11], (const float*)d_in[13],
                        (const float*)d_in[15]};
  const float* bd[3] = {(const float*)d_in[12], (const float*)d_in[14],
                        (const float*)d_in[16]};
  float* out = (float*)d_out;

  char* ws = (char*)d_ws;
  size_t off = 0;
  float* n_up = (float*)(ws + off);   off += (size_t)N_NEXUS * 192 * 4;
  float* n_out = (float*)(ws + off);  off += (size_t)N_NEXUS * 64 * 4;
  float* counts = (float*)(ws + off); off += (size_t)3 * N_PLANE * 4;
  _Float16* W1T = (_Float16*)(ws + off); off += (size_t)192 * 64 * 2;
  _Float16* W2T = (_Float16*)(ws + off); off += (size_t)64 * 64 * 2;
  _Float16* WdT[3];
  for (int p = 0; p < 3; ++p) { WdT[p] = (_Float16*)(ws + off); off += (size_t)128 * 64 * 2; }

  hipMemsetAsync(n_up, 0, (size_t)N_NEXUS * 192 * 4, stream);
  hipMemsetAsync(counts, 0, (size_t)3 * N_PLANE * 4, stream);
  hipMemsetAsync(d_out, 0, (size_t)3 * N_PLANE * 64 * 4, stream);

  wt_f16_kernel<<<(192 * 64 + 255) / 256, 256, 0, stream>>>(W1, W1T, 192, 64);
  wt_f16_kernel<<<(64 * 64 + 255) / 256, 256, 0, stream>>>(W2, W2T, 64, 64);
  for (int p = 0; p < 3; ++p)
    wt_f16_kernel<<<(128 * 64 + 255) / 256, 256, 0, stream>>>(Wd[p], WdT[p], 128, 64);

  const float* xs[3] = {x_u, x_v, x_y};
  const int* es[3] = {edge_u, edge_v, edge_y};
  const long upThreads = (long)N_EDGE * 64;
  for (int p = 0; p < 3; ++p) {
    up_scatter_kernel<<<(int)((upThreads + 255) / 256), 256, 0, stream>>>(
        xs[p], es[p], n_up, p * 64);
    count_kernel<<<(N_EDGE + 255) / 256, 256, 0, stream>>>(
        es[p], counts + (size_t)p * N_PLANE);
  }

  nexus_mlp_kernel<<<(N_NEXUS / 16 + 3) / 4, 128, 0, stream>>>(
      n_up, W1T, b1, W2T, b2, n_out);

  for (int p = 0; p < 3; ++p)
    nexus_down_kernel<<<N_EDGE / 16 / 4, 128, 0, stream>>>(
        xs[p], n_out, es[p], WdT[p], bd[p], out + (size_t)p * N_PLANE * 64);

  const long finThreads = 3L * N_PLANE * 64;
  finalize_kernel<<<(int)((finThreads + 255) / 256), 256, 0, stream>>>(out, counts);
}
